// GptOssDecoderLayer_40759239639580
// MI455X (gfx1250) — matrix operand order repaired and checked by reading, not compiled, not run on hardware
//
#include <hip/hip_runtime.h>
#include <hip/hip_bf16.h>
#include <math.h>

typedef __bf16 bf16;
typedef __attribute__((ext_vector_type(16))) __bf16 v16bf;
typedef __attribute__((ext_vector_type(8)))  float  v8f;

#define T_TOK   1024
#define HDIM    2880
#define NHEADS  64
#define NKVH    8
#define DHEAD   64
#define NEXP    16
#define IDIM    1440
#define QKVOUT  5120
#define QDIM    4096   // NHEADS*DHEAD
#define KVDIM   512    // NKVH*DHEAD
#define QTILES  (T_TOK/16)

union V16U { uint4 q[2]; v16bf v; };

static __device__ __forceinline__ v8f wmma_bf16(v16bf a, v16bf b, v8f c) {
  return __builtin_amdgcn_wmma_f32_16x16x32_bf16(false, a, false, b, (short)0, c, false, false);
}
static __device__ __forceinline__ v16bf load_frag16(const bf16* p) {
  V16U u;
  u.q[0] = *(const uint4*)p;
  u.q[1] = *(const uint4*)(p + 8);
  return u.v;
}
// CDNA5 async global->LDS copy (16B per lane), tracked by ASYNCcnt.
static __device__ __forceinline__ void async_copy_b128(unsigned lds_off, const void* gaddr) {
  asm volatile("global_load_async_to_lds_b128 %0, %1, off"
               :: "v"(lds_off), "v"(gaddr) : "memory");
}
static __device__ __forceinline__ void wait_asynccnt0() {
  asm volatile("s_wait_asynccnt 0x0" ::: "memory");
}

// ---------------------------------------------------------------- rmsnorm
__global__ __launch_bounds__(256)
void rmsnorm_kernel(const float* __restrict__ x, const float* __restrict__ w,
                    bf16* __restrict__ outb, float* __restrict__ outf) {
  int t = blockIdx.x, tid = threadIdx.x;
  const float* row = x + (long long)t * HDIM;
  float s = 0.f;
  for (int h = tid; h < HDIM; h += 256) { float v = row[h]; s += v * v; }
  for (int d = 16; d >= 1; d >>= 1) s += __shfl_xor(s, d);
  __shared__ float red[8];
  if ((tid & 31) == 0) red[tid >> 5] = s;
  __syncthreads();
  if (tid < 8) {
    float v = red[tid];
    for (int d = 4; d >= 1; d >>= 1) v += __shfl_xor(v, d);
    if (tid == 0) red[0] = v;
  }
  __syncthreads();
  float inv = rsqrtf(red[0] / (float)HDIM + 1e-6f);
  for (int h = tid; h < HDIM; h += 256) {
    float v = row[h] * inv * w[h];
    outb[(long long)t * HDIM + h] = (bf16)v;
    if (outf) outf[(long long)t * HDIM + h] = v;
  }
}

// ---------------------------------------------------------------- generic WMMA GEMM
// C[M,N] = A(bf16)[M,K] * B + bias ; transB=1: B[k][n]=W[n*ldw+k] ; transB=0: B[k][n]=W[k*ldw+n]
// mode 0: Cf[m][n]=val(+resid) ; mode 1: MoE activation -> actOut ; mode 2: Cf rows via c_rowmap
__global__ __launch_bounds__(256)
void gemm_kernel(const bf16* __restrict__ A, long long lda, long long a_zstride,
                 const int* __restrict__ a_rowmap,
                 const float* __restrict__ W, long long ldw, long long w_zstride, int transB,
                 const float* __restrict__ bias, long long bias_zstride,
                 const int* __restrict__ Mdyn, int M, int N, int K,
                 int mode,
                 float* __restrict__ Cf, long long ldc,
                 const int* __restrict__ c_rowmap,
                 bf16* __restrict__ actOut, long long act_ld,
                 const float* __restrict__ resid)
{
  int z  = blockIdx.z;
  int m0 = blockIdx.y * 128;
  int n0 = blockIdx.x * 128;
  int Mcur = Mdyn ? Mdyn[z] : M;
  if (m0 >= Mcur) return;
  bool fulltile = (m0 + 128 <= Mcur);   // uniform per block

  const bf16*  Ab   = A + (long long)z * a_zstride;
  const int*   rmap = a_rowmap ? (a_rowmap + (long long)z * T_TOK) : nullptr;
  const float* Wb   = W + (long long)z * w_zstride;
  const float* bb   = bias + (long long)z * bias_zstride;
  const int*   crm  = c_rowmap ? (c_rowmap + (long long)z * T_TOK) : nullptr;

  // A staged PRE-SWIZZLED into fragment order: Asw[mtile][lane][slot]
  __shared__ bf16 Asw[8][32][16];     // 8 KB
  __shared__ bf16 Bs[32][136];        // [k][n] padded, 8.7 KB

  int tid  = threadIdx.x;
  int wid  = tid >> 5, lane = tid & 31;
  int wm   = wid >> 1, wn = wid & 1;

  v8f zero8 = {0.f,0.f,0.f,0.f,0.f,0.f,0.f,0.f};
  v8f acc[2][4];
  for (int i = 0; i < 2; ++i) for (int j = 0; j < 4; ++j) acc[i][j] = zero8;

  // per-thread A staging geometry (8 contiguous bf16 -> one swizzled 16B slot run)
  int rrow = tid >> 2;
  int kc   = (tid & 3) * 8;            // {0,8,16,24}
  int hh   = (kc >> 3) & 1;            // dest half-wave
  int sb   = (kc >> 4) * 8;            // dest slot base {0,8}

  for (int k0 = 0; k0 < K; k0 += 32) {
    if (k0 + 32 < K)
      __builtin_prefetch(Wb + (long long)(transB ? (long long)n0 * ldw + (k0 + 32)
                                                 : (long long)(k0 + 32) * ldw + n0), 0, 1);
    __syncthreads();
    if (fulltile) {
      // CDNA5 async copy: global -> LDS directly, no VGPR round-trip
      for (int half = 0; half < 2; ++half) {
        int m = rrow + half * 64;
        long long ar = rmap ? (long long)rmap[m0 + m] : (long long)(m0 + m);
        const bf16* gsrc = Ab + ar * lda + k0 + kc;
        unsigned loff = (unsigned)(size_t)&Asw[m >> 4][(m & 15) + 16 * hh][sb];
        async_copy_b128(loff, gsrc);
      }
    } else {
      for (int half = 0; half < 2; ++half) {
        int m  = rrow + half * 64;
        int gm = m0 + m;
        uint4 val = make_uint4(0u, 0u, 0u, 0u);
        if (gm < Mcur) {
          long long ar = rmap ? (long long)rmap[gm] : (long long)gm;
          val = *(const uint4*)(Ab + ar * lda + k0 + kc);
        }
        *(uint4*)&Asw[m >> 4][(m & 15) + 16 * hh][sb] = val;
      }
    }
    if (transB) { // W is NxK: read along K, store transposed Bs[k][n]
      for (int it = 0; it < 4; ++it) {
        int chunk = tid + it * 256;
        int n  = chunk >> 3;
        int kq = (chunk & 7) * 4;
        int gn = n0 + n;
        float4 v = make_float4(0.f,0.f,0.f,0.f);
        if (gn < N) v = *(const float4*)(Wb + (long long)gn * ldw + k0 + kq);
        Bs[kq+0][n] = (bf16)v.x; Bs[kq+1][n] = (bf16)v.y;
        Bs[kq+2][n] = (bf16)v.z; Bs[kq+3][n] = (bf16)v.w;
      }
    } else {      // W is KxN: read along N
      for (int it = 0; it < 4; ++it) {
        int chunk = tid + it * 256;
        int kk = chunk >> 5;
        int nq = (chunk & 31) * 4;
        int gn = n0 + nq;
        float4 v = make_float4(0.f,0.f,0.f,0.f);
        if (gn + 3 < N) {
          v = *(const float4*)(Wb + (long long)(k0 + kk) * ldw + gn);
        } else {
          float tv[4] = {0.f,0.f,0.f,0.f};
          for (int q = 0; q < 4; ++q) if (gn + q < N) tv[q] = Wb[(long long)(k0 + kk) * ldw + gn + q];
          v = make_float4(tv[0],tv[1],tv[2],tv[3]);
        }
        Bs[kk][nq+0] = (bf16)v.x; Bs[kk][nq+1] = (bf16)v.y;
        Bs[kk][nq+2] = (bf16)v.z; Bs[kk][nq+3] = (bf16)v.w;
      }
    }
    if (fulltile) wait_asynccnt0();   // A-tile async copies landed in LDS
    __syncthreads();

    v16bf afr[2];
    afr[0] = load_frag16(&Asw[wm * 2 + 0][lane][0]);
    afr[1] = load_frag16(&Asw[wm * 2 + 1][lane][0]);
    for (int nt = 0; nt < 4; ++nt) {
      v16bf bfr;
      for (int s = 0; s < 16; ++s)
        bfr[s] = Bs[(s & 7) + 8 * (lane >> 4) + 16 * (s >> 3)][wn * 64 + nt * 16 + (lane & 15)];
      acc[0][nt] = wmma_bf16(afr[0], bfr, acc[0][nt]);
      acc[1][nt] = wmma_bf16(afr[1], bfr, acc[1][nt]);
    }
  }

  // epilogue; C layout: col = lane&15, row = r + 8*(lane>>4)
  int ln = lane & 15, hf = lane >> 4;
  for (int mt = 0; mt < 2; ++mt)
    for (int nt = 0; nt < 4; ++nt)
      for (int r = 0; r < 8; ++r) {
        int m = m0 + wm * 32 + mt * 16 + r + 8 * hf;
        int n = n0 + wn * 64 + nt * 16 + ln;
        bool ok = (m < Mcur) && (n < N);
        float val = ok ? (acc[mt][nt][r] + bb[n]) : 0.f;
        if (mode == 1) {
          float other = __shfl_xor(val, 1);    // gate/lin pair (even/odd n)
          if (ok && ((n & 1) == 0)) {
            float g  = fminf(val, 7.0f);
            float li = fminf(fmaxf(other, -7.0f), 7.0f);
            float a  = (li + 1.f) * (g / (1.f + __expf(-1.702f * g)));
            actOut[((long long)z * T_TOK + m) * act_ld + (n >> 1)] = (bf16)a;
          }
        } else if (ok) {
          long long orow = crm ? (long long)crm[m] : (long long)m;
          float outv = val;
          if (resid) outv += resid[(long long)m * ldc + n];
          Cf[orow * ldc + n] = outv;
        }
      }
}

// ---------------------------------------------------------------- RoPE (YaRN); Q written in
// WMMA-A fragment order: qswz[((h*QTILES + tq)*2 + half)*512 + lane*16 + slot]
__global__ __launch_bounds__(256)
void rope_kernel(const float* __restrict__ qkv, const int* __restrict__ pos_ids,
                 bf16* __restrict__ qswz, bf16* __restrict__ ktb, bf16* __restrict__ vbb)
{
  int t = blockIdx.x, tid = threadIdx.x;
  __shared__ float cs[32], sn[32];
  if (tid < 32) {
    int i = tid;
    float inv_extra = powf(150000.f, -(float)i / 32.f);
    float inv_inter = inv_extra * (1.f / 32.f);
    float ramp = fminf(fmaxf(((float)i - 8.f) * 0.1f, 0.f), 1.f); // low=8, high=18
    float ext  = 1.f - ramp;
    float invf = inv_inter * (1.f - ext) + inv_extra * ext;
    float ang  = (float)pos_ids[t] * invf;
    const float msc = 1.3465735903f; // 0.1*ln(32)+1
    cs[i] = cosf(ang) * msc;
    sn[i] = sinf(ang) * msc;
  }
  __syncthreads();
  const float* row = qkv + (long long)t * QKVOUT;
  int tq = t >> 4, mm = t & 15;
  for (int idx = tid; idx < NHEADS * 32; idx += 256) {
    int hh = idx >> 5, i = idx & 31;
    float x1 = row[hh * 64 + i], x2 = row[hh * 64 + 32 + i];
    float o1 = x1 * cs[i] - x2 * sn[i];   // dim i      (K-half 0)
    float o2 = x2 * cs[i] + x1 * sn[i];   // dim i + 32 (K-half 1)
    int lanep = mm + 16 * ((i >> 3) & 1);
    int slot  = (i & 7) + 8 * (i >> 4);
    long long base = (((long long)hh * QTILES + tq) * 2) * 512 + lanep * 16 + slot;
    qswz[base]       = (bf16)o1;
    qswz[base + 512] = (bf16)o2;
  }
  for (int idx = tid; idx < NKVH * 32; idx += 256) {
    int hh = idx >> 5, i = idx & 31;
    float x1 = row[QDIM + hh * 64 + i], x2 = row[QDIM + hh * 64 + 32 + i];
    ktb[(long long)(hh * 64 + i) * T_TOK + t]      = (bf16)(x1 * cs[i] - x2 * sn[i]);
    ktb[(long long)(hh * 64 + 32 + i) * T_TOK + t] = (bf16)(x2 * cs[i] + x1 * sn[i]);
  }
  for (int j = tid; j < KVDIM; j += 256)
    vbb[(long long)t * KVDIM + j] = (bf16)row[QDIM + KVDIM + j];
}

// ---------------------------------------------------------------- flash attention with sinks
__global__ __launch_bounds__(128)
void attn_kernel(const bf16* __restrict__ qswz, const bf16* __restrict__ ktb,
                 const bf16* __restrict__ vbb, const float* __restrict__ sinks,
                 bf16* __restrict__ attnb)
{
  int h = blockIdx.x, qblk = blockIdx.y;
  int wid = threadIdx.x >> 5, lane = threadIdx.x & 31;
  int q0  = qblk * 64 + wid * 16;
  int kvh = h >> 3;                    // rep = NHEADS/NKVH = 8
  __shared__ bf16 Plsw[4][32][16];     // per-wave P in fragment order

  v16bf aq[2];
  {
    const bf16* qbase = qswz + (((long long)h * QTILES + (q0 >> 4)) * 2) * 512 + lane * 16;
    aq[0] = load_frag16(qbase);
    aq[1] = load_frag16(qbase + 512);
  }

  v8f zero8 = {0.f,0.f,0.f,0.f,0.f,0.f,0.f,0.f};
  v8f acc[4]; for (int j = 0; j < 4; ++j) acc[j] = zero8;
  float mrow[8], lrow[8];
  for (int r = 0; r < 8; ++r) { mrow[r] = -INFINITY; lrow[r] = 0.f; }

  int ln = lane & 15, hf = lane >> 4;
  int kmax = q0 + 15;
  for (int kb0 = 0; kb0 <= kmax; kb0 += 32) {
    v8f s[2];
    for (int tile = 0; tile < 2; ++tile) {
      v8f zacc = zero8;
      for (int half = 0; half < 2; ++half) {
        v16bf bk;
        for (int s = 0; s < 16; ++s) {
          int kd = half * 32 + (s & 7) + 8 * (lane >> 4) + 16 * (s >> 3);
          bk[s] = ktb[(long long)(kvh * DHEAD + kd) * T_TOK + kb0 + tile * 16 + (lane & 15)];
        }
        zacc = wmma_bf16(aq[half], bk, zacc);
      }
      s[tile] = zacc;
    }
    float p0[8], p1[8];
    for (int r = 0; r < 8; ++r) {
      int mq = q0 + r + 8 * hf;
      bool v0 = (kb0 + ln) <= mq, v1 = (kb0 + 16 + ln) <= mq;
      float s0 = v0 ? s[0][r] * 0.125f : -INFINITY;
      float s1 = v1 ? s[1][r] * 0.125f : -INFINITY;
      float mx = fmaxf(s0, s1);
      for (int d = 1; d < 16; d <<= 1) mx = fmaxf(mx, __shfl_xor(mx, d));
      float mnew  = fmaxf(mrow[r], mx);
      float alpha = (mnew == -INFINITY) ? 1.f : __expf(mrow[r] - mnew);
      float e0 = v0 ? __expf(s0 - mnew) : 0.f;
      float e1 = v1 ? __expf(s1 - mnew) : 0.f;
      float rs = e0 + e1;
      for (int d = 1; d < 16; d <<= 1) rs += __shfl_xor(rs, d);
      lrow[r] = lrow[r] * alpha + rs;
      mrow[r] = mnew;
      for (int j = 0; j < 4; ++j) acc[j][r] *= alpha;
      p0[r] = e0; p1[r] = e1;
    }
    // C layout -> fragment-order LDS: k=ln -> slot ln&7 ; k=16+ln -> slot (ln&7)+8 ;
    // dest lane = m + 16*((ln>>3)&1)
    for (int r = 0; r < 8; ++r) {
      int m  = r + 8 * hf;
      int lp = m + 16 * ((ln >> 3) & 1);
      int sl = ln & 7;
      Plsw[wid][lp][sl]     = (bf16)p0[r];
      Plsw[wid][lp][sl + 8] = (bf16)p1[r];
    }
    v16bf ap = load_frag16(&Plsw[wid][lane][0]);
    for (int j = 0; j < 4; ++j) {
      v16bf bv;
      for (int s = 0; s < 16; ++s)
        bv[s] = vbb[(long long)(kb0 + (s & 7) + 8 * (lane >> 4) + 16 * (s >> 3)) * KVDIM + kvh * DHEAD + j * 16 + (lane & 15)];
      acc[j] = wmma_bf16(ap, bv, acc[j]);
    }
  }
  float snk = sinks[h];
  float inv[8];
  for (int r = 0; r < 8; ++r) { lrow[r] += __expf(snk - mrow[r]); inv[r] = 1.f / lrow[r]; }
  for (int j = 0; j < 4; ++j)
    for (int r = 0; r < 8; ++r) {
      int mq = q0 + r + 8 * hf;
      attnb[(long long)mq * QDIM + h * DHEAD + j * 16 + ln] = (bf16)(acc[j][r] * inv[r]);
    }
}

// ---------------------------------------------------------------- router / top-k
__global__ void zero_counts_kernel(int* c) { if (threadIdx.x < NEXP) c[threadIdx.x] = 0; }

__global__ __launch_bounds__(256)
void router_kernel(const float* __restrict__ xn, const float* __restrict__ rw,
                   const float* __restrict__ rb, int* __restrict__ counts,
                   int* __restrict__ token_list, int* __restrict__ pair_list,
                   float* __restrict__ wslot)
{
  int t = blockIdx.x, tid = threadIdx.x;
  int e = tid & 15, g = tid >> 4;
  const float* row = xn + (long long)t * HDIM;
  float p = 0.f;
  for (int h = g; h < HDIM; h += 16) p += row[h] * rw[e * HDIM + h];
  __shared__ float part[16][17];
  __shared__ float logits[16];
  part[g][e] = p;
  __syncthreads();
  if (tid < 16) {
    float s = rb[tid];
    for (int gg = 0; gg < 16; ++gg) s += part[gg][tid];
    logits[tid] = s;
  }
  __syncthreads();
  if (tid == 0) {
    float mx = logits[0];
    for (int j = 1; j < 16; ++j) mx = fmaxf(mx, logits[j]);
    float pr[16]; float den = 0.f;
    for (int j = 0; j < 16; ++j) { pr[j] = __expf(logits[j] - mx); den += pr[j]; }
    float iden = 1.f / den;
    for (int j = 0; j < 16; ++j) pr[j] *= iden;
    int idx[4]; float vals[4]; unsigned used = 0;
    for (int s2 = 0; s2 < 4; ++s2) {
      int best = 0; float bv = -1.f;
      for (int j = 0; j < 16; ++j)
        if (!((used >> j) & 1u) && pr[j] > bv) { bv = pr[j]; best = j; }
      used |= 1u << best; idx[s2] = best; vals[s2] = bv;
    }
    float s4 = vals[0] + vals[1] + vals[2] + vals[3];
    for (int s2 = 0; s2 < 4; ++s2) {
      int ee = idx[s2];
      int pos = atomicAdd(&counts[ee], 1);
      token_list[ee * T_TOK + pos] = t;
      pair_list[ee * T_TOK + pos] = t * 4 + s2;
      wslot[t * 4 + s2] = vals[s2] / s4;
    }
  }
}

// ---------------------------------------------------------------- final combine
__global__ __launch_bounds__(256)
void combine_kernel(const float* __restrict__ hbuf, const float* __restrict__ down_buf,
                    const float* __restrict__ wslot, float* __restrict__ out)
{
  int t = blockIdx.x, tid = threadIdx.x;
  float w0 = wslot[t*4+0], w1 = wslot[t*4+1], w2 = wslot[t*4+2], w3 = wslot[t*4+3];
  const float* d0 = down_buf + (long long)(t*4+0) * HDIM;
  const float* d1 = down_buf + (long long)(t*4+1) * HDIM;
  const float* d2 = down_buf + (long long)(t*4+2) * HDIM;
  const float* d3 = down_buf + (long long)(t*4+3) * HDIM;
  for (int n = tid; n < HDIM; n += 256)
    out[(long long)t * HDIM + n] = hbuf[(long long)t * HDIM + n]
        + w0 * d0[n] + w1 * d1[n] + w2 * d2[n] + w3 * d3[n];
}

// ---------------------------------------------------------------- host
extern "C" void kernel_launch(void* const* d_in, const int* in_sizes, int n_in,
                              void* d_out, int out_size, void* d_ws, size_t ws_size,
                              hipStream_t stream) {
  const float* hidden   = (const float*)d_in[0];
  const int*   pos_ids  = (const int*)  d_in[1];
  const float* ln1_w    = (const float*)d_in[2];
  const float* ln2_w    = (const float*)d_in[3];
  const float* w_qkv    = (const float*)d_in[4];
  const float* b_qkv    = (const float*)d_in[5];
  const float* w_o      = (const float*)d_in[6];
  const float* b_o      = (const float*)d_in[7];
  const float* sinks    = (const float*)d_in[8];
  const float* router_w = (const float*)d_in[9];
  const float* router_b = (const float*)d_in[10];
  const float* w_up     = (const float*)d_in[11];
  const float* b_up     = (const float*)d_in[12];
  const float* w_down   = (const float*)d_in[13];
  const float* b_down   = (const float*)d_in[14];
  float* out = (float*)d_out;

  size_t off = 0;
  auto take = [&](size_t bytes) -> char* {
    char* p = (char*)d_ws + off;
    off += (bytes + 255) & ~(size_t)255;
    return p;
  };
  float* qkv      = (float*)take((size_t)T_TOK * QKVOUT * 4);
  bf16*  xn1b     = (bf16*) take((size_t)T_TOK * HDIM * 2);
  bf16*  qswz     = (bf16*) take((size_t)T_TOK * QDIM * 2);
  bf16*  ktb      = (bf16*) take((size_t)KVDIM * T_TOK * 2);
  bf16*  vbb      = (bf16*) take((size_t)T_TOK * KVDIM * 2);
  bf16*  attnb    = (bf16*) take((size_t)T_TOK * QDIM * 2);
  float* hbuf     = (float*)take((size_t)T_TOK * HDIM * 4);
  float* xn2      = (float*)take((size_t)T_TOK * HDIM * 4);
  bf16*  xn2b     = (bf16*) take((size_t)T_TOK * HDIM * 2);
  int*   counts   = (int*)  take((size_t)NEXP * 4);
  int*   tok_list = (int*)  take((size_t)NEXP * T_TOK * 4);
  int*   pair_lst = (int*)  take((size_t)NEXP * T_TOK * 4);
  float* wslot    = (float*)take((size_t)T_TOK * 4 * 4);
  bf16*  actb     = (bf16*) take((size_t)NEXP * T_TOK * IDIM * 2);
  float* down_buf = (float*)take((size_t)T_TOK * 4 * HDIM * 4);

  rmsnorm_kernel<<<T_TOK, 256, 0, stream>>>(hidden, ln1_w, xn1b, nullptr);
  gemm_kernel<<<dim3(QKVOUT/128, T_TOK/128, 1), 256, 0, stream>>>(
      xn1b, HDIM, 0, nullptr, w_qkv, HDIM, 0, 1, b_qkv, 0,
      nullptr, T_TOK, QKVOUT, HDIM, 0, qkv, QKVOUT, nullptr, nullptr, 0, nullptr);
  rope_kernel<<<T_TOK, 256, 0, stream>>>(qkv, pos_ids, qswz, ktb, vbb);
  attn_kernel<<<dim3(NHEADS, T_TOK/64, 1), 128, 0, stream>>>(qswz, ktb, vbb, sinks, attnb);
  gemm_kernel<<<dim3((HDIM+127)/128, T_TOK/128, 1), 256, 0, stream>>>(
      attnb, QDIM, 0, nullptr, w_o, QDIM, 0, 1, b_o, 0,
      nullptr, T_TOK, HDIM, QDIM, 0, hbuf, HDIM, nullptr, nullptr, 0, hidden);
  rmsnorm_kernel<<<T_TOK, 256, 0, stream>>>(hbuf, ln2_w, xn2b, xn2);
  zero_counts_kernel<<<1, 32, 0, stream>>>(counts);
  router_kernel<<<T_TOK, 256, 0, stream>>>(xn2, router_w, router_b,
                                           counts, tok_list, pair_lst, wslot);
  gemm_kernel<<<dim3((2*IDIM+127)/128, T_TOK/128, NEXP), 256, 0, stream>>>(
      xn2b, HDIM, 0, tok_list, w_up, 2*IDIM, (long long)HDIM*2*IDIM, 0,
      b_up, 2*IDIM, counts, T_TOK, 2*IDIM, HDIM, 1,
      nullptr, 0, nullptr, actb, IDIM, nullptr);
  gemm_kernel<<<dim3((HDIM+127)/128, T_TOK/128, NEXP), 256, 0, stream>>>(
      actb, IDIM, (long long)T_TOK*IDIM, nullptr, w_down, HDIM, (long long)IDIM*HDIM, 0,
      b_down, HDIM, counts, T_TOK, HDIM, IDIM, 2,
      down_buf, HDIM, pair_lst, nullptr, 0, nullptr);
  combine_kernel<<<T_TOK, 256, 0, stream>>>(hbuf, down_buf, wslot, out);
}
